// SparseMoeBlock_88287347736703
// MI455X (gfx1250) — matrix operand order repaired and checked by reading, not compiled, not run on hardware
//
#include <hip/hip_runtime.h>

// ---------------- problem constants (match reference) ----------------
#define TB   8192      // tokens = B*S
#define HD   2048      // hidden
#define FF   1024      // ffn
#define NE   8         // experts
#define TPK  2         // top_k
#define TKS  (TB*TPK)  // total routed slots = 16384
#define BKK  64        // K-slab staged in LDS per GEMM stage

typedef __attribute__((ext_vector_type(8)))  float  v8f;
typedef __attribute__((ext_vector_type(16))) __bf16 v16bf;
typedef __attribute__((ext_vector_type(8)))  __bf16 v8bf;
typedef __attribute__((ext_vector_type(4)))  unsigned u32x4;
typedef __attribute__((ext_vector_type(8)))  int      i32x8;
typedef __attribute__((ext_vector_type(4)))  int      i32x4;

union Frag { v16bf v; v8bf h[2]; };

// float -> bf16, round-to-nearest-even
__device__ __forceinline__ __bf16 f2bf(float f) {
    union { float f; unsigned u; } in; in.f = f;
    unsigned r = in.u + 0x7fffu + ((in.u >> 16) & 1u);
    union { unsigned short s; __bf16 b; } out; out.s = (unsigned short)(r >> 16);
    return out.b;
}

// one 16-byte async global->LDS copy (tracked by ASYNCcnt)
__device__ __forceinline__ void async_cp16(const __bf16* gsrc, __bf16* ldst) {
    unsigned lds_off = (unsigned)(size_t)ldst;    // low 32 bits of generic = LDS offset
    asm volatile("global_load_async_to_lds_b128 %0, %1, off"
                 :: "v"(lds_off), "v"(gsrc) : "memory");
}
__device__ __forceinline__ void async_wait0() {
    asm volatile("s_wait_asynccnt 0x0" ::: "memory");
}

// TDM: move a 64x64 bf16 tile (row pitch = rowstride elems) into LDS with
// 16B per-row padding (LDS row stride = 128+16 bytes = BKK+8 elems).
// D# layout per CDNA5 ISA 8.3/8.4. Issue from one wave; tracked by TENSORcnt.
__device__ __forceinline__ void tdm_load_tile64(const __bf16* gsrc, unsigned lds_off,
                                                unsigned rowstride_elems) {
    size_t ga = (size_t)gsrc;
    u32x4 g0;
    g0[0] = 1u;                                        // count=1, user descriptor
    g0[1] = lds_off;                                   // lds_addr (bytes)
    g0[2] = (unsigned)(ga & 0xffffffffu);              // global_addr[31:0]
    g0[3] = (unsigned)((ga >> 32) & 0x1ffffffu)        // global_addr[56:32]
          | (2u << 30);                                // type=2 ("image")
    i32x8 g1;
    g1[0] = (1 << 16)                                  // data_size=1 -> 2 bytes
          | (1 << 20)                                  // pad_enable
          | (4 << 22)                                  // pad_interval: 32 DWORDs (=128B row)
          | (3 << 25);                                 // pad_amount: 4 DWORDs (=16B)
    g1[1] = (BKK << 16);                               // tensor_dim0 = 64 (lo16)
    g1[2] = (64 << 16);                                // tensor_dim0 hi=0 | tensor_dim1=64 lo
    g1[3] = (BKK << 16);                               // tensor_dim1 hi=0 | tile_dim0=64
    g1[4] = 64;                                        // tile_dim1=64, tile_dim2=0 (2D)
    g1[5] = (int)rowstride_elems;                      // tensor_dim0_stride lo32
    g1[6] = 0;                                         // stride hi | dim1_stride lo (unused)
    g1[7] = 0;
    i32x4 z4 = {0, 0, 0, 0};
    i32x8 z8 = {0, 0, 0, 0, 0, 0, 0, 0};
    // 6-arg form (amdgpu-toolchain / clang-23): (g0, g1, g2, g3, extra, cpol)
    __builtin_amdgcn_tensor_load_to_lds(g0, g1, z4, z4, z8, 0);
}
__device__ __forceinline__ void tensor_wait0() {
    __builtin_amdgcn_s_wait_tensorcnt(0);
}

// ---------------- 1) conversion fp32 -> bf16 (weights + activations) --------
__global__ void k_convert(const float* __restrict__ w13, const float* __restrict__ w2,
                          const float* __restrict__ x,
                          __bf16* __restrict__ w13b, __bf16* __restrict__ w2b,
                          __bf16* __restrict__ xb) {
    const size_t n13 = (size_t)NE * 2 * FF * HD / 4;   // float4 units
    const size_t n2  = (size_t)NE * HD * FF / 4;
    const size_t nx  = (size_t)TB * HD / 4;
    const size_t stride = (size_t)gridDim.x * blockDim.x;
    for (size_t i = (size_t)blockIdx.x * blockDim.x + threadIdx.x;
         i < n13 + n2 + nx; i += stride) {
        float4 v;
        __bf16* dst;
        if (i < n13)            { v = ((const float4*)w13)[i];            dst = w13b + i * 4; }
        else if (i < n13 + n2)  { v = ((const float4*)w2)[i - n13];       dst = w2b + (i - n13) * 4; }
        else                    { v = ((const float4*)x)[i - n13 - n2];   dst = xb + (i - n13 - n2) * 4; }
        dst[0] = f2bf(v.x); dst[1] = f2bf(v.y); dst[2] = f2bf(v.z); dst[3] = f2bf(v.w);
    }
}

// ---------------- 2) init counters ----------------
__global__ void k_init(int* __restrict__ counts, int* __restrict__ cursor) {
    int i = threadIdx.x;
    if (i < NE) { counts[i] = 0; cursor[i] = 0; }
}

// ---------------- 3) router: logits, softmax, top2 ----------------
// one wave32 per token; 8 waves per block
__global__ void k_router(const float* __restrict__ x, const float* __restrict__ gw,
                         float* __restrict__ logits, int* __restrict__ sel,
                         float* __restrict__ wts, int* __restrict__ counts) {
    const int wave = threadIdx.x >> 5;
    const int lane = threadIdx.x & 31;
    const int t = blockIdx.x * 8 + wave;
    if (t >= TB) return;

    float acc[NE];
#pragma unroll
    for (int e = 0; e < NE; ++e) acc[e] = 0.f;

    const float* xr = x + (size_t)t * HD;
    for (int h = lane; h < HD; h += 32) {
        float xv = xr[h];
#pragma unroll
        for (int e = 0; e < NE; ++e) acc[e] += xv * gw[e * HD + h];
    }
#pragma unroll
    for (int e = 0; e < NE; ++e) {
#pragma unroll
        for (int off = 16; off > 0; off >>= 1)
            acc[e] += __shfl_xor(acc[e], off, 32);
    }
    if (lane == 0) {
        float mx = acc[0];
#pragma unroll
        for (int e = 1; e < NE; ++e) mx = fmaxf(mx, acc[e]);
        float p[NE]; float sum = 0.f;
#pragma unroll
        for (int e = 0; e < NE; ++e) { p[e] = __expf(acc[e] - mx); sum += p[e]; }
        float inv = 1.f / sum;
#pragma unroll
        for (int e = 0; e < NE; ++e) { p[e] *= inv; logits[(size_t)t * NE + e] = acc[e]; }
        // top-2 (first index wins ties, like lax.top_k)
        int e0 = 0; float m0 = p[0];
#pragma unroll
        for (int e = 1; e < NE; ++e) if (p[e] > m0) { m0 = p[e]; e0 = e; }
        int e1 = -1; float m1 = -1.f;
#pragma unroll
        for (int e = 0; e < NE; ++e) if (e != e0 && p[e] > m1) { m1 = p[e]; e1 = e; }
        float norm = 1.f / (m0 + m1);
        sel[t * 2 + 0] = e0; wts[t * 2 + 0] = m0 * norm;
        sel[t * 2 + 1] = e1; wts[t * 2 + 1] = m1 * norm;
        atomicAdd(&counts[e0], 1);
        atomicAdd(&counts[e1], 1);
    }
}

// ---------------- 4) exclusive prefix over 8 experts ----------------
__global__ void k_bases(const int* __restrict__ counts, int* __restrict__ basev) {
    if (threadIdx.x == 0) {
        int acc = 0;
        for (int e = 0; e < NE; ++e) { basev[e] = acc; acc += counts[e]; }
    }
}

// ---------------- 5) compact slot assignment ----------------
__global__ void k_assign(const int* __restrict__ sel, const float* __restrict__ wts,
                         const int* __restrict__ basev, int* __restrict__ cursor,
                         int* __restrict__ slot2tok, float* __restrict__ slotwt) {
    int t = blockIdx.x * blockDim.x + threadIdx.x;
    if (t >= TB) return;
#pragma unroll
    for (int k = 0; k < TPK; ++k) {
        int e = sel[t * 2 + k];
        int pos = atomicAdd(&cursor[e], 1);
        int s = basev[e] + pos;
        slot2tok[s] = (t << 1) | k;
        slotwt[s] = wts[t * 2 + k];
    }
}

// ---------------- 6) GEMM1: h13 = x @ w13^T, fused SwiGLU -> inter (bf16) ----
// block = 256 thr (8 waves as 2x4), tile M=32 slots x N=64 gate feats (+ paired up)
// double-buffered LDS; weight tiles via TDM, activation gather via async-to-LDS
__global__ void k_gemm1(const __bf16* __restrict__ xb, const __bf16* __restrict__ w13b,
                        const int* __restrict__ slot2tok, const int* __restrict__ basev,
                        const int* __restrict__ counts, __bf16* __restrict__ inter) {
    const int e = blockIdx.z;
    const int cnt = counts[e];
    const int mTile = blockIdx.y;
    if (mTile * 32 >= cnt) return;
    const int nTile = blockIdx.x;                  // 16 tiles of 64 over FF
    const int s0 = basev[e] + mTile * 32;
    const int remaining = cnt - mTile * 32;        // 1..32 valid rows

    __shared__ __align__(16) __bf16 As[2][32][BKK + 8];
    __shared__ __align__(16) __bf16 Bg[2][64][BKK + 8];
    __shared__ __align__(16) __bf16 Bu[2][64][BKK + 8];
    __shared__ int toks[32];

    const int tid = threadIdx.x;
    if (tid < 32) {
        int r = (tid < remaining) ? tid : 0;
        toks[tid] = slot2tok[s0 + r] >> 1;         // token id
    }
    __syncthreads();

    const int wave = tid >> 5, lane = tid & 31;
    const int moff = (wave & 1) * 16;
    const int noff = (wave >> 1) * 16;

    v8f cg = {}; v8f cu = {};

    const __bf16* wg = w13b + ((size_t)e * 2 * FF + nTile * 64) * HD;
    const __bf16* wu = w13b + ((size_t)e * 2 * FF + FF + nTile * 64) * HD;

    // B tiles (gate + up): one TDM descriptor each, issued by wave 0
    auto stageB = [&](int k0, int b) {
        if (tid < 32) {
            tdm_load_tile64(wg + k0, (unsigned)(size_t)&Bg[b][0][0], HD);
            tdm_load_tile64(wu + k0, (unsigned)(size_t)&Bu[b][0][0], HD);
        }
    };
    // A tile: gathered token rows (bf16), async copies
    auto stageA = [&](int k0, int b) {
        for (int i = tid; i < 32 * (BKK / 8); i += 256) {
            int r = i >> 3, c8 = (i & 7) * 8;
            async_cp16(&xb[(size_t)toks[r] * HD + k0 + c8], &As[b][r][c8]);
        }
    };

    stageB(0, 0);
    stageA(0, 0);
    async_wait0();
    if (tid < 32) tensor_wait0();
    __syncthreads();

    const int NSTEP = HD / BKK;
    for (int step = 0; step < NSTEP; ++step) {
        const int cur = step & 1;
        if (step + 1 < NSTEP) {                    // prefetch next slab into alt buffer
            stageB((step + 1) * BKK, cur ^ 1);
            stageA((step + 1) * BKK, cur ^ 1);
        }
#pragma unroll
        for (int kk = 0; kk < BKK; kk += 32) {
            // A fragment (contiguous 16 K per lane half, matching B)
            const int ca = kk + ((lane >> 4) << 4);
            const int ar = moff + (lane & 15);
            Frag a, bg, bu;
            a.h[0] = *(const v8bf*)&As[cur][ar][ca];
            a.h[1] = *(const v8bf*)&As[cur][ar][ca + 8];
            // B fragment (lane = column, contiguous 16 K per lane half)
            const int cb = kk + ((lane >> 4) << 4);
            const int br = noff + (lane & 15);
            bg.h[0] = *(const v8bf*)&Bg[cur][br][cb]; bg.h[1] = *(const v8bf*)&Bg[cur][br][cb + 8];
            bu.h[0] = *(const v8bf*)&Bu[cur][br][cb]; bu.h[1] = *(const v8bf*)&Bu[cur][br][cb + 8];
            cg = __builtin_amdgcn_wmma_f32_16x16x32_bf16(false, a.v, false, bg.v,
                                                         (short)0, cg, false, false);
            cu = __builtin_amdgcn_wmma_f32_16x16x32_bf16(false, a.v, false, bu.v,
                                                         (short)0, cu, false, false);
        }
        async_wait0();                             // A slab landed
        if (tid < 32) tensor_wait0();              // B slabs landed (wave 0's TENSORcnt)
        __syncthreads();
    }
    // epilogue: SwiGLU, store bf16 inter[slot][f]
    const int mbase = moff + ((lane >> 4) << 3);
    const int ncol = nTile * 64 + noff + (lane & 15);
#pragma unroll
    for (int i = 0; i < 8; ++i) {
        int m = mbase + i;
        if (m < remaining) {
            float g = cg[i], u = cu[i];
            float sig = 1.f / (1.f + __expf(-g));
            inter[(size_t)(s0 + m) * FF + ncol] = f2bf(g * sig * u);
        }
    }
}

// ---------------- 7) GEMM2: y = inter @ w2^T, weighted scatter to out2 ------
__global__ void k_gemm2(const __bf16* __restrict__ inter, const __bf16* __restrict__ w2b,
                        const int* __restrict__ slot2tok, const int* __restrict__ basev,
                        const int* __restrict__ counts, const float* __restrict__ slotwt,
                        float* __restrict__ out2) {
    const int e = blockIdx.z;
    const int cnt = counts[e];
    const int mTile = blockIdx.y;
    if (mTile * 32 >= cnt) return;
    const int hTile = blockIdx.x;                  // 32 tiles of 64 over HD
    const int s0 = basev[e] + mTile * 32;
    const int remaining = cnt - mTile * 32;

    __shared__ __align__(16) __bf16 As[2][32][BKK + 8];
    __shared__ __align__(16) __bf16 Bs[2][64][BKK + 8];

    const int tid = threadIdx.x;
    const int wave = tid >> 5, lane = tid & 31;
    const int moff = (wave & 1) * 16;
    const int noff = (wave >> 1) * 16;

    v8f c = {};
    const __bf16* wb = w2b + ((size_t)e * HD + hTile * 64) * FF;

    auto stage = [&](int k0, int b) {
        // B tile via TDM (wave 0)
        if (tid < 32)
            tdm_load_tile64(wb + k0, (unsigned)(size_t)&Bs[b][0][0], FF);
        // A tile: inter rows (bf16), async copies; clamp rows past 'remaining'
        for (int i = tid; i < 32 * (BKK / 8); i += 256) {
            int r = i >> 3, c8 = (i & 7) * 8;
            int rr = (r < remaining) ? r : 0;
            async_cp16(&inter[(size_t)(s0 + rr) * FF + k0 + c8], &As[b][r][c8]);
        }
    };

    stage(0, 0);
    async_wait0();
    if (tid < 32) tensor_wait0();
    __syncthreads();

    const int NSTEP = FF / BKK;
    for (int step = 0; step < NSTEP; ++step) {
        const int cur = step & 1;
        if (step + 1 < NSTEP) stage((step + 1) * BKK, cur ^ 1);
#pragma unroll
        for (int kk = 0; kk < BKK; kk += 32) {
            const int ca = kk + ((lane >> 4) << 4);
            const int ar = moff + (lane & 15);
            Frag a, b;
            a.h[0] = *(const v8bf*)&As[cur][ar][ca];
            a.h[1] = *(const v8bf*)&As[cur][ar][ca + 8];
            const int cb = kk + ((lane >> 4) << 4);
            const int br = noff + (lane & 15);
            b.h[0] = *(const v8bf*)&Bs[cur][br][cb]; b.h[1] = *(const v8bf*)&Bs[cur][br][cb + 8];
            c = __builtin_amdgcn_wmma_f32_16x16x32_bf16(false, a.v, false, b.v,
                                                        (short)0, c, false, false);
        }
        async_wait0();
        if (tid < 32) tensor_wait0();
        __syncthreads();
    }
    const int mbase = moff + ((lane >> 4) << 3);
    const int hcol = hTile * 64 + noff + (lane & 15);
#pragma unroll
    for (int i = 0; i < 8; ++i) {
        int m = mbase + i;
        if (m < remaining) {
            int s = s0 + m;
            int tk = slot2tok[s];                  // t*2 + k
            out2[(size_t)tk * HD + hcol] = c[i] * slotwt[s];
        }
    }
}

// ---------------- 8) combine: out[t] = out2[2t] + out2[2t+1] ----------------
__global__ void k_combine(const float* __restrict__ out2, float* __restrict__ out) {
    const size_t n4 = (size_t)TB * HD / 4;
    const size_t stride = (size_t)gridDim.x * blockDim.x;
    for (size_t i = (size_t)blockIdx.x * blockDim.x + threadIdx.x; i < n4; i += stride) {
        size_t t = i / (HD / 4);
        size_t h4 = i % (HD / 4);
        float4 a = ((const float4*)(out2 + (t * 2 + 0) * HD))[h4];
        float4 b = ((const float4*)(out2 + (t * 2 + 1) * HD))[h4];
        float4 r; r.x = a.x + b.x; r.y = a.y + b.y; r.z = a.z + b.z; r.w = a.w + b.w;
        ((float4*)(out + t * HD))[h4] = r;
    }
}

// ---------------- launcher ----------------
// workspace layout (requires ~302 MB):
//   w13b  bf16 [8][2048][2048]   67,108,864 B
//   w2b   bf16 [8][2048][1024]   33,554,432 B
//   xb    bf16 [8192][2048]      33,554,432 B
//   inter bf16 [16384][1024]     33,554,432 B
//   out2  f32  [16384][2048]    134,217,728 B
//   sel i32[16384], wts f32[16384], slot2tok i32[16384], slotwt f32[16384]
//   counts/base/cursor i32[8] each
extern "C" void kernel_launch(void* const* d_in, const int* in_sizes, int n_in,
                              void* d_out, int out_size, void* d_ws, size_t ws_size,
                              hipStream_t stream) {
    const float* x      = (const float*)d_in[0];
    const float* gate_w = (const float*)d_in[1];
    const float* w13    = (const float*)d_in[2];
    const float* w2     = (const float*)d_in[3];

    float* out    = (float*)d_out;
    float* logits = out + (size_t)TB * HD;

    char* p = (char*)d_ws;
    __bf16* w13b   = (__bf16*)p;  p += (size_t)NE * 2 * FF * HD * 2;
    __bf16* w2b    = (__bf16*)p;  p += (size_t)NE * HD * FF * 2;
    __bf16* xb     = (__bf16*)p;  p += (size_t)TB * HD * 2;
    __bf16* inter  = (__bf16*)p;  p += (size_t)TKS * FF * 2;
    float*  out2   = (float*)p;   p += (size_t)TKS * HD * 4;
    int*    sel    = (int*)p;     p += (size_t)TKS * 4;
    float*  wts    = (float*)p;   p += (size_t)TKS * 4;
    int*    s2t    = (int*)p;     p += (size_t)TKS * 4;
    float*  swt    = (float*)p;   p += (size_t)TKS * 4;
    int*    counts = (int*)p;     p += 64;
    int*    basev  = (int*)p;     p += 64;
    int*    cursor = (int*)p;     p += 64;

    k_convert<<<8192, 256, 0, stream>>>(w13, w2, x, w13b, w2b, xb);
    k_init<<<1, 64, 0, stream>>>(counts, cursor);
    k_router<<<TB / 8, 256, 0, stream>>>(x, gate_w, logits, sel, wts, counts);
    k_bases<<<1, 32, 0, stream>>>(counts, basev);
    k_assign<<<TB / 256, 256, 0, stream>>>(sel, wts, basev, cursor, s2t, swt);
    k_gemm1<<<dim3(FF / 64, TB / 32, NE), 256, 0, stream>>>(xb, w13b, s2t, basev, counts, inter);
    k_gemm2<<<dim3(HD / 64, TB / 32, NE), 256, 0, stream>>>(inter, w2b, s2t, basev, counts, swt, out2);
    k_combine<<<4096, 256, 0, stream>>>(out2, out);
}
